// MPDRLActor_56435870270045
// MI455X (gfx1250) — matrix operand order repaired and checked by reading, not compiled, not run on hardware
//
#include <hip/hip_runtime.h>
#include <hip/hip_bf16.h>
#include <math.h>

typedef __attribute__((ext_vector_type(16))) _Float16 v16h;
typedef __attribute__((ext_vector_type(8)))  float    v8f;

// ---------------- problem constants ----------------
constexpr int   E_   = 320000;   // edge-nodes
constexpr int   F_   = 20;       // features
// packed-weight offsets (in _Float16 units). Each fragment = 512 halves
// (32 lanes x 16 halves), stored so lane L loads v16h at frag*32 + L.
constexpr int OFF_PW1  = 0;      // msg_W1   : 2 kc * 2 nc
constexpr int OFF_PW2  = 2048;   // msg_W2   : 2 nc
constexpr int OFF_PWIH = 3072;   // gru_Wih^T: 3 gate * 2 nc
constexpr int OFF_PWHH = 6144;   // gru_Whh^T: 3 gate * 2 nc
constexpr int OFF_PB1  = 9216;   // ro_W1    : 4 nc
constexpr int OFF_PB2  = 11264;  // ro_W2    : 2 kc * 2 nc
constexpr int OFF_PB3  = 13312;  // ro_W3    : 1 (col 0 only)
// workspace byte offsets
constexpr size_t WS_H_OFF   = 32768;                       // E*20 f32
constexpr size_t WS_AGG_OFF = WS_H_OFF + (size_t)E_*F_*4;  // E*20 f32

// ---------------- helpers ----------------
static __device__ __forceinline__ v8f wmma16(v16h a, v16h b, v8f c) {
  return __builtin_amdgcn_wmma_f32_16x16x32_f16(
      /*neg_a=*/false, a, /*neg_b=*/false, b,
      /*c_mod=*/(short)0, c, /*reuse_a=*/false, /*reuse_b=*/false);
}
// Branchless SELU on the native TRANS pipe (v_exp_f32 + v_cndmask).
static __device__ __forceinline__ float selu_f(float x) {
  const float a = 1.6732632423543772f, s = 1.0507009873554805f;
  float e = __expf(fminf(x, 0.f));       // always evaluated, no exec branching
  float neg = s * a * (e - 1.f);
  return x > 0.f ? s * x : neg;
}
// sigmoid via v_exp_f32 + v_rcp_f32 (no IEEE-refined divide)
static __device__ __forceinline__ float sigm_f(float x) {
  return __builtin_amdgcn_rcpf(1.f + __expf(-x));
}
// gfx1250 native v_tanh_f32 when available
static __device__ __forceinline__ float tanh_f(float x) {
#if __has_builtin(__builtin_amdgcn_tanhf)
  return __builtin_amdgcn_tanhf(x);
#else
  float e = __expf(2.f * x);
  return 1.f - 2.f * __builtin_amdgcn_rcpf(e + 1.f);
#endif
}

// ---------------- weight packing (runs once) ----------------
// B-fragment layout (f16 32x16): lane L, element i -> K = (i&7) + 16*(i/8) + 8*(L/16),
// N = L%16. Value stored at frag*512 + L*16 + i.
__global__ void pack_weights(const float* __restrict__ W1,   // 40x32
                             const float* __restrict__ W2,   // 32x20
                             const float* __restrict__ Wih,  // 60x20
                             const float* __restrict__ Whh,  // 60x20
                             const float* __restrict__ rW1,  // 20x64
                             const float* __restrict__ rW2,  // 64x32
                             const float* __restrict__ rW3,  // 32x1
                             _Float16* __restrict__ wp) {
  int idx = threadIdx.x;
  if (idx >= 512) return;
  int lane = idx >> 4, i = idx & 15;
  int Kb  = (i & 7) + ((i >> 3) << 4) + ((lane >> 4) << 3);  // 0..31
  int N16 = lane & 15;
  int slot = lane * 16 + i;
  // msg_W1 (40x32): kc in {0,1}, nc in {0,1}
  for (int kc = 0; kc < 2; ++kc)
    for (int nc = 0; nc < 2; ++nc) {
      int K = Kb + kc * 32, N = nc * 16 + N16;
      float v = (K < 40) ? W1[K * 32 + N] : 0.f;
      wp[OFF_PW1 + (kc * 2 + nc) * 512 + slot] = (_Float16)v;
    }
  // msg_W2 (32x20): nc in {0,1}
  for (int nc = 0; nc < 2; ++nc) {
    int N = nc * 16 + N16;
    float v = (N < 20) ? W2[Kb * 20 + N] : 0.f;
    wp[OFF_PW2 + nc * 512 + slot] = (_Float16)v;
  }
  // gru: B[k][g*20+j] = W[(g*20+j)*20 + k]  (effective Wih^T / Whh^T)
  for (int g = 0; g < 3; ++g)
    for (int nc = 0; nc < 2; ++nc) {
      int j = nc * 16 + N16;
      float vi = 0.f, vh = 0.f;
      if (Kb < 20 && j < 20) {
        vi = Wih[(g * 20 + j) * 20 + Kb];
        vh = Whh[(g * 20 + j) * 20 + Kb];
      }
      wp[OFF_PWIH + (g * 2 + nc) * 512 + slot] = (_Float16)vi;
      wp[OFF_PWHH + (g * 2 + nc) * 512 + slot] = (_Float16)vh;
    }
  // ro_W1 (20x64): nc in {0..3}
  for (int nc = 0; nc < 4; ++nc) {
    int N = nc * 16 + N16;
    float v = (Kb < 20) ? rW1[Kb * 64 + N] : 0.f;
    wp[OFF_PB1 + nc * 512 + slot] = (_Float16)v;
  }
  // ro_W2 (64x32): kc,nc in {0,1}
  for (int kc = 0; kc < 2; ++kc)
    for (int nc = 0; nc < 2; ++nc) {
      int K = Kb + kc * 32, N = nc * 16 + N16;
      wp[OFF_PB2 + (kc * 2 + nc) * 512 + slot] = (_Float16)rW2[K * 32 + N];
    }
  // ro_W3 (32x1) -> column 0 only
  wp[OFF_PB3 + slot] = (_Float16)((N16 == 0) ? rW3[Kb] : 0.f);
}

// ---------------- h init ----------------
__global__ void init_h(const float* __restrict__ src, float* __restrict__ dst, int n) {
  int i = blockIdx.x * blockDim.x + threadIdx.x;
  if (i < n) dst[i] = src[i];
}

// ---------------- message MLP + segment sum (one wave per edge) ----------------
__global__ __launch_bounds__(128) void msg_agg_kernel(
    const float* __restrict__ h, const int* __restrict__ edge_index,
    const float* __restrict__ b1, const float* __restrict__ b2,
    const _Float16* __restrict__ wp, float* __restrict__ agg) {
  __shared__ float    hif[4][20];    // broadcast h_i row
  __shared__ float    hjf[4][320];   // 16 contiguous h_j rows
  __shared__ _Float16 hid[4][512];   // selu(hidden) 16x32 f16, repack C->A
  const int wave = threadIdx.x >> 5, lane = threadIdx.x & 31;
  const int e = blockIdx.x * 4 + wave;
  const int dst = edge_index[E_ + e];                  // dst node of edge e
  const long nb = (long)dst * 16 * F_;                 // nbr block base (floats)
  if (lane < F_) hif[wave][lane] = h[(long)e * F_ + lane];
#pragma unroll
  for (int q = 0; q < 10; ++q) hjf[wave][q * 32 + lane] = h[nb + q * 32 + lane];
  __syncthreads();

  const int M = lane & 15, hg = lane >> 4;
  v16h a0, a1;
#pragma unroll
  for (int i = 0; i < 16; ++i) {
    int K = (i & 7) + ((i >> 3) << 4) + (hg << 3);     // A layout K (chunk 0)
    float v0 = (K < F_) ? hif[wave][K] : hjf[wave][M * F_ + (K - F_)];
    int K1 = K + 32;                                   // chunk 1 (pad 40->64)
    float v1 = (K1 < 2 * F_) ? hjf[wave][M * F_ + (K1 - F_)] : 0.f;
    a0[i] = (_Float16)v0;
    a1[i] = (_Float16)v1;
  }
  const v16h* PW1 = (const v16h*)(wp + OFF_PW1);
  const v16h* PW2 = (const v16h*)(wp + OFF_PW2);

  const float b1v0 = b1[lane & 15], b1v1 = b1[16 + (lane & 15)];
  v8f c0, c1;
#pragma unroll
  for (int r = 0; r < 8; ++r) { c0[r] = b1v0; c1[r] = b1v1; }
  c0 = wmma16(a0, PW1[0 * 32 + lane], c0);   // kc0,nc0
  c0 = wmma16(a1, PW1[2 * 32 + lane], c0);   // kc1,nc0
  c1 = wmma16(a0, PW1[1 * 32 + lane], c1);   // kc0,nc1
  c1 = wmma16(a1, PW1[3 * 32 + lane], c1);   // kc1,nc1
#pragma unroll
  for (int r = 0; r < 8; ++r) {
    int Mr = r + (hg << 3);
    hid[wave][Mr * 32 + (lane & 15)]      = (_Float16)selu_f(c0[r]);
    hid[wave][Mr * 32 + 16 + (lane & 15)] = (_Float16)selu_f(c1[r]);
  }
  __syncthreads();

  v16h a2;
#pragma unroll
  for (int i = 0; i < 16; ++i) {
    int K = (i & 7) + ((i >> 3) << 4) + (hg << 3);
    a2[i] = hid[wave][M * 32 + K];
  }
  const int col0 = lane & 15, col1 = 16 + (lane & 15);
  v8f d0, d1;
  const float b20 = b2[col0], b21 = (col1 < F_) ? b2[col1] : 0.f;
#pragma unroll
  for (int r = 0; r < 8; ++r) { d0[r] = b20; d1[r] = b21; }
  d0 = wmma16(a2, PW2[0 * 32 + lane], d0);
  d1 = wmma16(a2, PW2[1 * 32 + lane], d1);

  // segment-sum over the 16 rows (== the 16 messages of edge e)
  float s0 = 0.f, s1 = 0.f;
#pragma unroll
  for (int r = 0; r < 8; ++r) { s0 += d0[r]; s1 += d1[r]; }
  s0 += __shfl_xor(s0, 16, 32);
  s1 += __shfl_xor(s1, 16, 32);
  if (lane < 16) {
    agg[(long)e * F_ + col0] = s0;
    if (col1 < F_) agg[(long)e * F_ + col1] = s1;
  }
}

// ---------------- GRU update (one wave per 16-edge tile, in-place h) ----------------
__global__ __launch_bounds__(128) void gru_kernel(
    float* __restrict__ h, const float* __restrict__ agg,
    const float* __restrict__ bih, const float* __restrict__ bhh,
    const _Float16* __restrict__ wp) {
  __shared__ float af[4][320];
  __shared__ float hf[4][320];
  const int wave = threadIdx.x >> 5, lane = threadIdx.x & 31;
  const int t = blockIdx.x * 4 + wave;
  const long base = (long)t * 16 * F_;
#pragma unroll
  for (int q = 0; q < 10; ++q) {
    af[wave][q * 32 + lane] = agg[base + q * 32 + lane];
    hf[wave][q * 32 + lane] = h[base + q * 32 + lane];
  }
  __syncthreads();

  const int M = lane & 15, hg = lane >> 4;
  v16h aa, ah;
#pragma unroll
  for (int i = 0; i < 16; ++i) {
    int K = (i & 7) + ((i >> 3) << 4) + (hg << 3);
    aa[i] = (_Float16)((K < F_) ? af[wave][M * F_ + K] : 0.f);
    ah[i] = (_Float16)((K < F_) ? hf[wave][M * F_ + K] : 0.f);
  }
  const v16h* PIH = (const v16h*)(wp + OFF_PWIH);
  const v16h* PHH = (const v16h*)(wp + OFF_PWHH);

  v8f ci[3][2], ch[3][2];
#pragma unroll
  for (int g = 0; g < 3; ++g)
#pragma unroll
    for (int nc = 0; nc < 2; ++nc) {
      int j = nc * 16 + (lane & 15);
      float bi = (j < F_) ? bih[g * F_ + j] : 0.f;
      float bh = (j < F_) ? bhh[g * F_ + j] : 0.f;
      v8f ti, th;
#pragma unroll
      for (int r = 0; r < 8; ++r) { ti[r] = bi; th[r] = bh; }
      ti = wmma16(aa, PIH[(g * 2 + nc) * 32 + lane], ti);
      th = wmma16(ah, PHH[(g * 2 + nc) * 32 + lane], th);
      ci[g][nc] = ti; ch[g][nc] = th;
    }
  // gates land in identical lane/element positions -> purely element-wise
#pragma unroll
  for (int nc = 0; nc < 2; ++nc) {
    int col = nc * 16 + (lane & 15);
    bool valid = col < F_;
#pragma unroll
    for (int r = 0; r < 8; ++r) {
      float rg = sigm_f(ci[0][nc][r] + ch[0][nc][r]);
      float zg = sigm_f(ci[1][nc][r] + ch[1][nc][r]);
      float ng = tanh_f(ci[2][nc][r] + rg * ch[2][nc][r]);
      int Mr = r + (hg << 3);
      float hold = valid ? hf[wave][Mr * F_ + col] : 0.f;
      float hnew = (1.f - zg) * ng + zg * hold;
      if (valid) h[base + (long)Mr * F_ + col] = hnew;
    }
  }
}

// ---------------- readout MLP (one wave per 16-row tile) ----------------
__global__ __launch_bounds__(128) void readout_kernel(
    const float* __restrict__ h, const float* __restrict__ b1,
    const float* __restrict__ b2, const float* __restrict__ b3,
    const _Float16* __restrict__ wp, float* __restrict__ out) {
  __shared__ float    hf[4][320];
  __shared__ _Float16 x1[4][1024];  // 16x64
  __shared__ _Float16 x2[4][512];   // 16x32
  const int wave = threadIdx.x >> 5, lane = threadIdx.x & 31;
  const int t = blockIdx.x * 4 + wave;
  const long base = (long)t * 16 * F_;
#pragma unroll
  for (int q = 0; q < 10; ++q) hf[wave][q * 32 + lane] = h[base + q * 32 + lane];
  __syncthreads();

  const int M = lane & 15, hg = lane >> 4;
  v16h a;
#pragma unroll
  for (int i = 0; i < 16; ++i) {
    int K = (i & 7) + ((i >> 3) << 4) + (hg << 3);
    a[i] = (_Float16)((K < F_) ? hf[wave][M * F_ + K] : 0.f);
  }
  const v16h* PB1 = (const v16h*)(wp + OFF_PB1);
#pragma unroll
  for (int nc = 0; nc < 4; ++nc) {
    int col = nc * 16 + (lane & 15);
    float bb = b1[col];
    v8f c;
#pragma unroll
    for (int r = 0; r < 8; ++r) c[r] = bb;
    c = wmma16(a, PB1[nc * 32 + lane], c);
#pragma unroll
    for (int r = 0; r < 8; ++r) {
      int Mr = r + (hg << 3);
      x1[wave][Mr * 64 + col] = (_Float16)selu_f(c[r]);
    }
  }
  __syncthreads();

  v16h a20, a21;
#pragma unroll
  for (int i = 0; i < 16; ++i) {
    int K = (i & 7) + ((i >> 3) << 4) + (hg << 3);
    a20[i] = x1[wave][M * 64 + K];
    a21[i] = x1[wave][M * 64 + K + 32];
  }
  const v16h* PB2 = (const v16h*)(wp + OFF_PB2);
#pragma unroll
  for (int nc = 0; nc < 2; ++nc) {
    int col = nc * 16 + (lane & 15);
    float bb = b2[col];
    v8f c;
#pragma unroll
    for (int r = 0; r < 8; ++r) c[r] = bb;
    c = wmma16(a20, PB2[(0 * 2 + nc) * 32 + lane], c);
    c = wmma16(a21, PB2[(1 * 2 + nc) * 32 + lane], c);
#pragma unroll
    for (int r = 0; r < 8; ++r) {
      int Mr = r + (hg << 3);
      x2[wave][Mr * 32 + col] = (_Float16)selu_f(c[r]);
    }
  }
  __syncthreads();

  v16h a3;
#pragma unroll
  for (int i = 0; i < 16; ++i) {
    int K = (i & 7) + ((i >> 3) << 4) + (hg << 3);
    a3[i] = x2[wave][M * 32 + K];
  }
  const v16h* PB3 = (const v16h*)(wp + OFF_PB3);
  const float bb3 = b3[0];
  v8f c;
#pragma unroll
  for (int r = 0; r < 8; ++r) c[r] = ((lane & 15) == 0) ? bb3 : 0.f;
  c = wmma16(a3, PB3[lane], c);
  if ((lane & 15) == 0) {
#pragma unroll
    for (int r = 0; r < 8; ++r) {
      int Mr = r + (hg << 3);
      float x = c[r];
      // softplus on the native exp/log pipe; exact for large x
      float w = (x > 20.f) ? x : __logf(1.f + __expf(x));
      w += 0.1f;
      w = fminf(fmaxf(w, 0.1f), 10.f);
      out[(long)t * 16 + Mr] = w;
    }
  }
}

// ---------------- host launch ----------------
extern "C" void kernel_launch(void* const* d_in, const int* in_sizes, int n_in,
                              void* d_out, int out_size, void* d_ws, size_t ws_size,
                              hipStream_t stream) {
  const float* edge_attr = (const float*)d_in[0];
  const float* msg_W1 = (const float*)d_in[1];
  const float* msg_b1 = (const float*)d_in[2];
  const float* msg_W2 = (const float*)d_in[3];
  const float* msg_b2 = (const float*)d_in[4];
  const float* gru_Wih = (const float*)d_in[5];
  const float* gru_Whh = (const float*)d_in[6];
  const float* gru_bih = (const float*)d_in[7];
  const float* gru_bhh = (const float*)d_in[8];
  const float* ro_W1 = (const float*)d_in[9];
  const float* ro_b1 = (const float*)d_in[10];
  const float* ro_W2 = (const float*)d_in[11];
  const float* ro_b2 = (const float*)d_in[12];
  const float* ro_W3 = (const float*)d_in[13];
  const float* ro_b3 = (const float*)d_in[14];
  const int* edge_index = (const int*)d_in[15];
  float* out = (float*)d_out;

  _Float16* wp = (_Float16*)d_ws;
  float* h   = (float*)((char*)d_ws + WS_H_OFF);
  float* agg = (float*)((char*)d_ws + WS_AGG_OFF);

  pack_weights<<<1, 512, 0, stream>>>(msg_W1, msg_W2, gru_Wih, gru_Whh,
                                      ro_W1, ro_W2, ro_W3, wp);
  const int nh = E_ * F_;
  init_h<<<(nh + 255) / 256, 256, 0, stream>>>(edge_attr, h, nh);

  const int msg_blocks = E_ / 4;        // one wave (of 4/block) per edge
  const int tile_blocks = E_ / 16 / 4;  // one wave per 16-row tile
  for (int step = 0; step < 4; ++step) {
    msg_agg_kernel<<<msg_blocks, 128, 0, stream>>>(h, edge_index, msg_b1,
                                                   msg_b2, wp, agg);
    gru_kernel<<<tile_blocks, 128, 0, stream>>>(h, agg, gru_bih, gru_bhh, wp);
  }
  readout_kernel<<<tile_blocks, 128, 0, stream>>>(h, ro_b1, ro_b2, ro_b3, wp, out);
}
